// RNN_71210557767829
// MI455X (gfx1250) — matrix operand order repaired and checked by reading, not compiled, not run on hardware
//
#include <hip/hip_runtime.h>
#include <hip/hip_bf16.h>
#include <math.h>

typedef __bf16 bf16;
typedef __attribute__((ext_vector_type(16))) bf16  v16bf;
typedef __attribute__((ext_vector_type(8)))  bf16  v8bf;
typedef __attribute__((ext_vector_type(8)))  float v8f;

#define WMMA_BF16(a, b, c) \
  __builtin_amdgcn_wmma_f32_16x16x32_bf16(false, (a), false, (b), (short)0, (c), false, false)

// ---------------------------------------------------------------------------
// Fragment loaders (wave32, contiguous k = 16*(lane/16) + s in A and B alike)
// A: 16x32 bf16 (MxK), row-major, lda in elements.
__device__ __forceinline__ v16bf load_a_frag(const bf16* __restrict__ A, int lda,
                                             int m0, int k0) {
  int lane = threadIdx.x & 31;
  const bf16* p = A + (size_t)(m0 + (lane & 15)) * lda + k0 + ((lane >> 4) << 4);
  v8bf lo = *(const v8bf*)p;
  v8bf hi = *(const v8bf*)(p + 8);
  return __builtin_shufflevector(lo, hi, 0,1,2,3,4,5,6,7,8,9,10,11,12,13,14,15);
}

// B: 32x16 bf16 (KxN) from TRANSPOSED weights Bt[n][k] in global memory.
__device__ __forceinline__ v16bf load_b_frag(const bf16* __restrict__ Bt, int ldb,
                                             int n0, int k0) {
  int lane = threadIdx.x & 31;
  const bf16* p = Bt + (size_t)(n0 + (lane & 15)) * ldb + k0 + ((lane >> 4) << 4);
  return *(const v16bf*)p;
}

// B fragment from an LDS-resident gate tile (row stride LDK elements, padded).
template <int LDK>
__device__ __forceinline__ v16bf load_b_frag_lds(const bf16* base, int k0) {
  int lane = threadIdx.x & 31;
  const bf16* p = base + (lane & 15) * LDK + k0 + ((lane >> 4) << 4);
  v8bf lo = *(const v8bf*)p;
  v8bf hi = *(const v8bf*)(p + 8);
  return __builtin_shufflevector(lo, hi, 0,1,2,3,4,5,6,7,8,9,10,11,12,13,14,15);
}

__device__ __forceinline__ float hsig(float x) {
  return fminf(fmaxf(0.2f * x + 0.5f, 0.0f), 1.0f);
}

__device__ __forceinline__ float fast_tanh(float x) {
#if __has_builtin(__builtin_amdgcn_tanhf)
  return __builtin_amdgcn_tanhf(x);
#elif __has_builtin(__builtin_amdgcn_tanh_f32)
  return __builtin_amdgcn_tanh_f32(x);
#else
  return tanhf(x);
#endif
}

// ---------------------------------------------------------------------------
// LDS-tiled transpose + f32->bf16: src [K=1024][N=4096] -> dst [N][K] bf16.
// Coalesced reads and writes; runs on every graph replay, so it must be cheap.
__global__ __launch_bounds__(256) void k_transpose_cvt(const float* __restrict__ src,
                                                       bf16* __restrict__ dst) {
  constexpr int K = 1024, N = 4096;
  __shared__ float tile[32][33];
  int tk = blockIdx.x & 31;   // K/32 = 32 tiles
  int tn = blockIdx.x >> 5;   // N/32 = 128 tiles
  int tx = threadIdx.x & 31;
  int ty = threadIdx.x >> 5;  // 8 rows per pass
  int k0 = tk << 5, n0 = tn << 5;
#pragma unroll
  for (int r = 0; r < 32; r += 8)
    tile[ty + r][tx] = src[(size_t)(k0 + ty + r) * N + n0 + tx];
  __syncthreads();
#pragma unroll
  for (int r = 0; r < 32; r += 8)
    dst[(size_t)(n0 + ty + r) * K + k0 + tx] = (bf16)tile[tx][ty + r];
}

__global__ __launch_bounds__(256) void k_f32_to_bf16(const float* __restrict__ src,
                                                     bf16* __restrict__ dst) {
  size_t i = ((size_t)blockIdx.x * 256 + threadIdx.x) * 8;
  float4 a = *(const float4*)(src + i);
  float4 b = *(const float4*)(src + i + 4);
  v8bf o;
  o[0] = (bf16)a.x; o[1] = (bf16)a.y; o[2] = (bf16)a.z; o[3] = (bf16)a.w;
  o[4] = (bf16)b.x; o[5] = (bf16)b.y; o[6] = (bf16)b.z; o[7] = (bf16)b.w;
  *(v8bf*)(dst + i) = o;
}

__global__ __launch_bounds__(256) void k_init(unsigned* bar, bf16* h0, bf16* h1) {
  if (threadIdx.x < 64) bar[threadIdx.x] = 0u;
  for (int j = threadIdx.x; j < 32 * 1024; j += 256) {
    h0[j] = (bf16)0.0f;
    h1[j] = (bf16)0.0f;
  }
}

// ---------------------------------------------------------------------------
// Input-projection GEMM: C[M,4096] = A[M,1024](bf16) @ B + bias (f32 out).
// WG tile 128x128, 8 waves, wave = 2 M-tiles x 4 N-tiles.
__global__ __launch_bounds__(256) void k_gemm_bias(const bf16* __restrict__ A,
                                                   const bf16* __restrict__ Bt,
                                                   const float* __restrict__ bias,
                                                   float* __restrict__ C) {
  constexpr int K = 1024, N = 4096;
  int lane = threadIdx.x & 31;
  int w    = threadIdx.x >> 5;
  int mbase = blockIdx.y * 128 + (w >> 1) * 32;
  int nbase = blockIdx.x * 128 + (w & 1) * 64;
  v8f acc[2][4] = {};
  for (int k0 = 0; k0 < K; k0 += 32) {
    v16bf a0 = load_a_frag(A, K, mbase,      k0);
    v16bf a1 = load_a_frag(A, K, mbase + 16, k0);
    v16bf b0 = load_b_frag(Bt, K, nbase,      k0);
    v16bf b1 = load_b_frag(Bt, K, nbase + 16, k0);
    v16bf b2 = load_b_frag(Bt, K, nbase + 32, k0);
    v16bf b3 = load_b_frag(Bt, K, nbase + 48, k0);
    acc[0][0] = WMMA_BF16(a0, b0, acc[0][0]);
    acc[0][1] = WMMA_BF16(a0, b1, acc[0][1]);
    acc[0][2] = WMMA_BF16(a0, b2, acc[0][2]);
    acc[0][3] = WMMA_BF16(a0, b3, acc[0][3]);
    acc[1][0] = WMMA_BF16(a1, b0, acc[1][0]);
    acc[1][1] = WMMA_BF16(a1, b1, acc[1][1]);
    acc[1][2] = WMMA_BF16(a1, b2, acc[1][2]);
    acc[1][3] = WMMA_BF16(a1, b3, acc[1][3]);
  }
  int nn = lane & 15, rsel = (lane >> 4) << 3;
#pragma unroll
  for (int i = 0; i < 2; ++i)
#pragma unroll
    for (int j = 0; j < 4; ++j) {
      int col = nbase + j * 16 + nn;
      float bz = bias[col];
#pragma unroll
      for (int r = 0; r < 8; ++r) {
        int row = mbase + i * 16 + rsel + r;
        C[(size_t)row * N + col] = acc[i][j][r] + bz;
      }
    }
}

// ---------------------------------------------------------------------------
// Persistent LSTM recurrence: 64 WGs x 8 waves; WG owns 16 h-columns.
// The WG's U slice (4 gate tiles x 16 cols x K=1024 bf16 = 128 KB) lives in
// LDS (padded rows -> conflict-free ds_read_b128). h ping-pongs through
// global (L2-resident). One grid barrier per timestep.
__global__ __launch_bounds__(256) void k_lstm_persistent(
    const float* __restrict__ xz, const bf16* __restrict__ Ut,
    bf16* h_buf0, bf16* h_buf1, bf16* seq_out, float* final_out,
    unsigned* bar, int T) {
  constexpr int H = 1024;
  constexpr int LDK = H + 8;  // padded row stride: 2064 B, banks 4l..4l+3
  constexpr unsigned NWG = 64;
  __shared__ alignas(32) bf16 u_sm[4][16][LDK];  // ~129 KB
  __shared__ float z_sm[4][32][16];
  __shared__ float c_sm[32][16];

  int wg   = blockIdx.x;
  int lane = threadIdx.x & 31;
  int w    = threadIdx.x >> 5;
  int m0   = (w & 1) << 4;   // batch tile base (0 / 16)
  int gate = w >> 1;         // 0=i 1=f 2=g 3=o
  int n0g  = wg << 4;        // h-column base

  // Stage U slice: 64 rows x 1024 bf16, coalesced 16B chunks.
  for (int chunk = threadIdx.x; chunk < 64 * (H / 8); chunk += 256) {
    int r  = chunk >> 7;         // /(H/8) = /128 -> 0..63
    int kc = (chunk & 127) << 3; // k offset in elements
    int g = r >> 4, n = r & 15;
    *(v8bf*)&u_sm[g][n][kc] =
        *(const v8bf*)(Ut + ((size_t)(g * H + n0g + n) << 10) + kc);
  }
  for (int i = threadIdx.x; i < 512; i += 256) ((float*)c_sm)[i] = 0.0f;
  unsigned local_gen = 0;
  __syncthreads();

  int nn = lane & 15;
  int rbase = m0 + ((lane >> 4) << 3);
  const bf16* ug = &u_sm[gate][0][0];

  for (int t = 0; t < T; ++t) {
    const bf16* hprev = (t & 1) ? h_buf1 : h_buf0;
    bf16*       hnext = (t & 1) ? h_buf0 : h_buf1;

    // Hoist xz loads (overlap with GEMM) + prefetch next step's lines.
    float xzv[8];
#pragma unroll
    for (int r = 0; r < 8; ++r) {
      size_t base = ((size_t)(rbase + r) * T + t) * 4096 + gate * H + n0g + nn;
      xzv[r] = xz[base];
      if (t + 1 < T) __builtin_prefetch(&xz[base + 4096], 0, 1);
    }

    // z-tile = h_prev @ U (32x16 for one gate); two accumulator chains.
    v8f acc0 = {}, acc1 = {};
#pragma unroll 4
    for (int k0 = 0; k0 < H; k0 += 64) {
      v16bf a0 = load_a_frag(hprev, H, m0, k0);
      v16bf b0 = load_b_frag_lds<LDK>(ug, k0);
      acc0 = WMMA_BF16(a0, b0, acc0);
      v16bf a1 = load_a_frag(hprev, H, m0, k0 + 32);
      v16bf b1 = load_b_frag_lds<LDK>(ug, k0 + 32);
      acc1 = WMMA_BF16(a1, b1, acc1);
    }

#pragma unroll
    for (int r = 0; r < 8; ++r)
      z_sm[gate][rbase + r][nn] = acc0[r] + acc1[r] + xzv[r];
    __syncthreads();

    // Fused gate math + cell update + h emit (512 elems / 256 threads).
    for (int idx = threadIdx.x; idx < 512; idx += 256) {
      int bi = idx >> 4, nl = idx & 15;
      float gi = hsig(z_sm[0][bi][nl]);
      float gf = hsig(z_sm[1][bi][nl]);
      float gg = fast_tanh(z_sm[2][bi][nl]);
      float go = hsig(z_sm[3][bi][nl]);
      float c  = gf * c_sm[bi][nl] + gi * gg;
      c_sm[bi][nl] = c;
      float h  = go * fast_tanh(c);
      int col = n0g + nl;
      hnext[bi * H + col] = (bf16)h;
      if (seq_out) seq_out[((size_t)bi * T + t) * H + col] = (bf16)h;
      if (final_out && t == T - 1) final_out[bi * H + col] = h;
    }
    __syncthreads();

    // Grid-wide barrier (counter + generation).
    __builtin_amdgcn_fence(__ATOMIC_RELEASE, "agent");
    if (threadIdx.x == 0) {
      unsigned prev = atomicAdd(bar, 1u);
      if (prev == NWG - 1) {
        atomicExch(bar, 0u);
        __builtin_amdgcn_fence(__ATOMIC_SEQ_CST, "agent");
        atomicAdd(bar + 1, 1u);
      } else {
        volatile unsigned* vg = bar + 1;
        while (*vg == local_gen) __builtin_amdgcn_s_sleep(1);
      }
    }
    local_gen++;
    __syncthreads();
    __builtin_amdgcn_fence(__ATOMIC_ACQUIRE, "agent");
  }
}

// ---------------------------------------------------------------------------
extern "C" void kernel_launch(void* const* d_in, const int* in_sizes, int n_in,
                              void* d_out, int out_size, void* d_ws, size_t ws_size,
                              hipStream_t stream) {
  (void)in_sizes; (void)n_in; (void)out_size; (void)ws_size;
  const float* x  = (const float*)d_in[0];
  const float* W0 = (const float*)d_in[1];
  const float* U0 = (const float*)d_in[2];
  const float* b0 = (const float*)d_in[3];
  const float* W1 = (const float*)d_in[4];
  const float* U1 = (const float*)d_in[5];
  const float* b1 = (const float*)d_in[6];
  float* out = (float*)d_out;

  constexpr size_t B = 32, T = 512, D = 1024, H = 1024, NG = 4096;

  char* p = (char*)d_ws;
  auto take = [&](size_t bytes) {
    char* r = p;
    p += (bytes + 255) & ~(size_t)255;
    return r;
  };
  unsigned* bar  = (unsigned*)take(256);
  bf16* xbf      = (bf16*)take(B * T * D * 2);
  bf16* h1seq    = (bf16*)take(B * T * H * 2);
  bf16* Wt0      = (bf16*)take(NG * H * 2);
  bf16* Ut0      = (bf16*)take(NG * H * 2);
  bf16* Wt1      = (bf16*)take(NG * H * 2);
  bf16* Ut1      = (bf16*)take(NG * H * 2);
  float* xz      = (float*)take(B * T * NG * 4);
  bf16* hb0      = (bf16*)take(B * H * 2);
  bf16* hb1      = (bf16*)take(B * H * 2);

  dim3 blk(256);
  k_transpose_cvt<<<4096, blk, 0, stream>>>(W0, Wt0);
  k_transpose_cvt<<<4096, blk, 0, stream>>>(U0, Ut0);
  k_transpose_cvt<<<4096, blk, 0, stream>>>(W1, Wt1);
  k_transpose_cvt<<<4096, blk, 0, stream>>>(U1, Ut1);
  k_f32_to_bf16<<<8192, blk, 0, stream>>>(x, xbf);

  // Layer 0
  k_init<<<1, blk, 0, stream>>>(bar, hb0, hb1);
  k_gemm_bias<<<dim3(32, 128), blk, 0, stream>>>(xbf, Wt0, b0, xz);
  k_lstm_persistent<<<64, blk, 0, stream>>>(xz, Ut0, hb0, hb1, h1seq, nullptr,
                                            bar, (int)T);
  // Layer 1
  k_init<<<1, blk, 0, stream>>>(bar, hb0, hb1);
  k_gemm_bias<<<dim3(32, 128), blk, 0, stream>>>(h1seq, Wt1, b1, xz);
  k_lstm_persistent<<<64, blk, 0, stream>>>(xz, Ut1, hb0, hb1, nullptr, out,
                                            bar, (int)T);
}
